// ChronosCore_31258771980627
// MI455X (gfx1250) — matrix operand order repaired and checked by reading, not compiled, not run on hardware
//
#include <hip/hip_runtime.h>
#include <hip/hip_fp16.h>

typedef float    v2f  __attribute__((ext_vector_type(2)));
typedef float    v8f  __attribute__((ext_vector_type(8)));
typedef _Float16 v16h __attribute__((ext_vector_type(16)));
typedef _Float16 v8h  __attribute__((ext_vector_type(8)));
typedef unsigned int u32x4 __attribute__((ext_vector_type(4)));
typedef int          i32x4 __attribute__((ext_vector_type(4)));
typedef int          i32x8 __attribute__((ext_vector_type(8)));

#define Bsz  4
#define Tlen 512
#define Dm   1024
#define Vv   32000
#define Pp   256
#define Ss   4096
#define KDm  64
#define VDm  64
#define Kk   4
#define Hm   1024
#define Lm   1024
#define MACW 1536   /* D + P + K*VD */

#define GBLK 64
#define NTHR 256
#define WPB  8
#define TW   (GBLK*WPB)

// lm-head tiling
#define LOG_YGRP   4          /* grid.y */
#define MT_PER_BLK 32         /* 128 M-tiles / LOG_YGRP */
#define MT_PER_WAVE 4         /* MT_PER_BLK / WPB */
#define BT_PITCH   1040       /* f16 LDS row pitch (pad 16 halves vs 1024) */

#if defined(__has_builtin)
#if __has_builtin(__builtin_amdgcn_tensor_load_to_lds) && \
    __has_builtin(__builtin_amdgcn_s_wait_tensorcnt)
#define USE_TDM 1
#endif
#endif

// ---------------- workspace layout (float units) ----------------
#define WS_X      0            /* 2,097,152 : x[B,T,D]      */
#define WS_ENCALL 2097152      /* 2,097,152 : enc_all[B,T,D]*/
#define WS_YH     4194304      /* 1,048,576 : y f16 [2048,1024] */
#define WS_ZZ     5242880      /* zeroed zone start */
#define Z_XT    0
#define Z_Q     16384
#define Z_SIM   17408
#define Z_MAC   82944
#define Z_ENC   107520
#define Z_H     123904
#define Z_L     140288
#define Z_LINP  156672
#define Z_GI    189440
#define Z_GH    238592
#define Z_GIL   287744
#define Z_GHL   336896
#define Z_FLAGS 386048
#define Z_TOTAL 386112

// output offsets (float units)
#define OUT_LOGITS 0
#define OUT_VALS   65536000
#define OUT_IDX    66060288

// ---------------- helpers ----------------
__device__ __forceinline__ void gbar(int* cnt, int* gen) {
  __syncthreads();
  if (threadIdx.x == 0) {
    __threadfence();
    int g = __hip_atomic_load(gen, __ATOMIC_RELAXED, __HIP_MEMORY_SCOPE_AGENT);
    int a = __hip_atomic_fetch_add(cnt, 1, __ATOMIC_ACQ_REL, __HIP_MEMORY_SCOPE_AGENT);
    if (a == (int)gridDim.x - 1) {
      __hip_atomic_store(cnt, 0, __ATOMIC_RELAXED, __HIP_MEMORY_SCOPE_AGENT);
      __hip_atomic_fetch_add(gen, 1, __ATOMIC_RELEASE, __HIP_MEMORY_SCOPE_AGENT);
    } else {
      while (__hip_atomic_load(gen, __ATOMIC_ACQUIRE, __HIP_MEMORY_SCOPE_AGENT) == g)
        __builtin_amdgcn_s_sleep(2);
    }
    __threadfence();
  }
  __syncthreads();
}

// f32 WMMA 16x16x4 K-loop: A[16,K] (lda), B = W[n0+0..15, 0..K-1]^T (ldw)
__device__ __forceinline__ v8f wmma_kloop_f32(const float* __restrict__ A, int lda,
                                              const float* __restrict__ W, int ldw,
                                              int Kdim, int lane) {
  const int r  = lane & 15;
  const int ko = (lane >> 4) << 1;
  const float* ap = A + r * lda + ko;
  const float* wp = W + r * ldw + ko;
  v8f acc = {};
  for (int k = 0; k < Kdim; k += 4) {
    v2f a = *(const v2f*)(ap + k);
    v2f b = *(const v2f*)(wp + k);
    __builtin_prefetch(wp + k + 128, 0, 1);
    acc = __builtin_amdgcn_wmma_f32_16x16x4_f32(false, a, false, b, (short)0, acc,
                                                false, false);
  }
  return acc;
}

__device__ __forceinline__ float sigm(float x) { return 1.f / (1.f + __expf(-x)); }
__device__ __forceinline__ float gru_out(float ir, float iz, float inn,
                                         float hr, float hz, float hn, float hv) {
  float r = sigm(ir + hr);
  float z = sigm(iz + hz);
  float n = tanhf(inn + r * hn);
  return (1.f - z) * n + z * hv;
}

// ---------------- kernels ----------------
__global__ void zero_kernel(float* p, int n) {
  int i = blockIdx.x * blockDim.x + threadIdx.x;
  if (i < n) p[i] = 0.f;
}

__global__ void embed_kernel(const int* __restrict__ ids,
                             const float* __restrict__ tok,
                             const float* __restrict__ pos,
                             float* __restrict__ x) {
  long i = (long)blockIdx.x * blockDim.x + threadIdx.x;
  if (i >= (long)Bsz * Tlen * Dm) return;
  int d = (int)(i & (Dm - 1));
  long bt = i >> 10;
  int t = (int)(bt & (Tlen - 1));
  int b = (int)(bt >> 9);
  x[i] = tok[(long)ids[b * Tlen + t] * Dm + d] + pos[t * Dm + d];
}

__device__ void topk_row(const float* __restrict__ sim, int row, int t,
                         float* __restrict__ mac,
                         const float* __restrict__ ltm_vals,
                         float* __restrict__ out_vals, int* __restrict__ out_idx,
                         float* sv, int* six) {
  const int tid = threadIdx.x;
  float bv[4]; int bi[4];
#pragma unroll
  for (int j = 0; j < 4; ++j) { bv[j] = -3.4e38f; bi[j] = 0x7fffffff; }
  for (int s = tid; s < Ss; s += NTHR) {
    float v = sim[row * Ss + s];
#pragma unroll
    for (int j = 0; j < 4; ++j) {
      if (v > bv[j] || (v == bv[j] && s < bi[j])) {
        for (int q2 = 3; q2 > j; --q2) { bv[q2] = bv[q2 - 1]; bi[q2] = bi[q2 - 1]; }
        bv[j] = v; bi[j] = s; break;
      }
    }
  }
#pragma unroll
  for (int j = 0; j < 4; ++j) { sv[tid * 4 + j] = bv[j]; six[tid * 4 + j] = bi[j]; }
  __syncthreads();
  for (int off = NTHR / 2; off > 0; off >>= 1) {
    if (tid < off) {
      float mv[4]; int mi[4];
#pragma unroll
      for (int j = 0; j < 4; ++j) { mv[j] = sv[tid * 4 + j]; mi[j] = six[tid * 4 + j]; }
#pragma unroll
      for (int j = 0; j < 4; ++j) {
        float v = sv[(tid + off) * 4 + j]; int s = six[(tid + off) * 4 + j];
#pragma unroll
        for (int p = 0; p < 4; ++p) {
          if (v > mv[p] || (v == mv[p] && s < mi[p])) {
            for (int q2 = 3; q2 > p; --q2) { mv[q2] = mv[q2 - 1]; mi[q2] = mi[q2 - 1]; }
            mv[p] = v; mi[p] = s; break;
          }
        }
      }
#pragma unroll
      for (int j = 0; j < 4; ++j) { sv[tid * 4 + j] = mv[j]; six[tid * 4 + j] = mi[j]; }
    }
    __syncthreads();
  }
  if (tid < Kk) out_idx[(row * Tlen + t) * Kk + tid] = six[tid];
  __syncthreads();
  {
    int ki = tid >> 6, d = tid & 63;           // 256 threads = 4*64
    int id = six[ki];
    float vv = ltm_vals[id * VDm + d];
    mac[row * MACW + Dm + Pp + ki * VDm + d] = vv;
    out_vals[((row * Tlen + t) * Kk + ki) * VDm + d] = vv;
  }
}

__global__ __launch_bounds__(NTHR) void chronos_persistent(
    const float* __restrict__ x, const float* __restrict__ persistent,
    const float* __restrict__ ltm_keys, const float* __restrict__ ltm_vals,
    const float* __restrict__ qproj_w,
    const float* __restrict__ in_proj_w, const float* __restrict__ in_proj_b,
    const float* __restrict__ h_Wi, const float* __restrict__ h_bi,
    const float* __restrict__ h_Wh, const float* __restrict__ h_bh,
    const float* __restrict__ h2c_w, const float* __restrict__ h2c_b,
    const float* __restrict__ l_Wi, const float* __restrict__ l_bi,
    const float* __restrict__ l_Wh, const float* __restrict__ l_bh,
    const float* __restrict__ l2o_w, const float* __restrict__ l2o_b,
    float* xt, float* q, float* sim, float* mac, float* enc,
    float* h, float* l, float* linp, float* gi, float* gh, float* gil, float* ghl,
    int* flags, float* enc_all, float* out_vals, int* out_idx) {
  __shared__ float sv[NTHR * 4];
  __shared__ int   six[NTHR * 4];
  const int lane = threadIdx.x & 31;
  const int widx = threadIdx.x >> 5;
  const int gw   = blockIdx.x * WPB + widx;
  int* cnt  = flags + 0;
  int* gen  = flags + 16;
  int* done = flags + 32;
  int* viol = flags + 48;
  const int r16 = lane & 15, mb0 = (lane >> 4) * 8;

  for (int t = 0; t < Tlen; ++t) {
    // P0: stage x[:,t,:] into padded xt
    for (int i = blockIdx.x * NTHR + threadIdx.x; i < Bsz * Dm; i += GBLK * NTHR) {
      int b = i >> 10, d = i & 1023;
      xt[b * Dm + d] = x[((long)b * Tlen + t) * Dm + d];
    }
    gbar(cnt, gen);
    // P1: query = xt @ qproj_w.T  [16,64]
    for (int tile = gw; tile < KDm / 16; tile += TW) {
      v8f acc = wmma_kloop_f32(xt, Dm, qproj_w + tile * 16 * Dm, Dm, Dm, lane);
      int n = tile * 16 + r16;
#pragma unroll
      for (int j = 0; j < 8; ++j) q[(mb0 + j) * KDm + n] = acc[j];
    }
    gbar(cnt, gen);
    // P2: sim = q @ ltm_keys.T  [16,4096]
    for (int tile = gw; tile < Ss / 16; tile += TW) {
      v8f acc = wmma_kloop_f32(q, KDm, ltm_keys + tile * 16 * KDm, KDm, KDm, lane);
      int n = tile * 16 + r16;
#pragma unroll
      for (int j = 0; j < 8; ++j) sim[(mb0 + j) * Ss + n] = acc[j];
    }
    gbar(cnt, gen);
    // P3: top-k + gather (blocks 0..3), mac token/persistent fill (others)
    if (blockIdx.x < Bsz) {
      topk_row(sim, blockIdx.x, t, mac, ltm_vals, out_vals, out_idx, sv, six);
    } else {
      for (int e = (blockIdx.x - Bsz) * NTHR + threadIdx.x; e < Bsz * (Dm + Pp);
           e += (GBLK - Bsz) * NTHR) {
        int b = e / (Dm + Pp), c = e % (Dm + Pp);
        mac[b * MACW + c] = (c < Dm) ? xt[b * Dm + c] : persistent[c - Dm];
      }
    }
    gbar(cnt, gen);
    // P4: enc = gelu(mac @ in_proj_w.T + b)
    for (int tile = gw; tile < Dm / 16; tile += TW) {
      v8f acc = wmma_kloop_f32(mac, MACW, in_proj_w + tile * 16 * MACW, MACW, MACW, lane);
      int n = tile * 16 + r16;
#pragma unroll
      for (int j = 0; j < 8; ++j) {
        float v = acc[j] + in_proj_b[n];
        enc[(mb0 + j) * Dm + n] = v * 0.5f * (1.f + erff(v * 0.70710678118654752f));
      }
    }
    gbar(cnt, gen);
    for (int hs = 0; hs < 2; ++hs) {
      // P5: gi = enc@h_Wi.T + h_bi ; gh = h@h_Wh.T + h_bh  (384 tiles)
      for (int tile = gw; tile < 384; tile += TW) {
        if (tile < 192) {
          v8f acc = wmma_kloop_f32(enc, Dm, h_Wi + tile * 16 * Dm, Dm, Dm, lane);
          int n = tile * 16 + r16;
#pragma unroll
          for (int j = 0; j < 8; ++j) gi[(mb0 + j) * 3 * Hm + n] = acc[j] + h_bi[n];
        } else {
          int tt = tile - 192;
          v8f acc = wmma_kloop_f32(h, Hm, h_Wh + tt * 16 * Hm, Hm, Hm, lane);
          int n = tt * 16 + r16;
#pragma unroll
          for (int j = 0; j < 8; ++j) gh[(mb0 + j) * 3 * Hm + n] = acc[j] + h_bh[n];
        }
      }
      gbar(cnt, gen);
      // P6: h GRU update, l_input[:, :D] = enc, reset convergence flags
      for (int i = blockIdx.x * NTHR + threadIdx.x; i < Bsz * Hm; i += GBLK * NTHR) {
        int b = i >> 10, n = i & 1023;
        float hv = h[b * Hm + n];
        float o = gru_out(gi[b * 3072 + n], gi[b * 3072 + 1024 + n], gi[b * 3072 + 2048 + n],
                          gh[b * 3072 + n], gh[b * 3072 + 1024 + n], gh[b * 3072 + 2048 + n], hv);
        h[b * Hm + n] = o;
        linp[b * 2048 + n] = enc[b * Dm + n];
      }
      if (blockIdx.x == 0 && threadIdx.x == 0) { *done = 0; *viol = 0; }
      gbar(cnt, gen);
      // P7: context = h@h2c_w.T + b  -> l_input[:, D:2D]
      for (int tile = gw; tile < Dm / 16; tile += TW) {
        v8f acc = wmma_kloop_f32(h, Hm, h2c_w + tile * 16 * Hm, Hm, Hm, lane);
        int n = tile * 16 + r16;
#pragma unroll
        for (int j = 0; j < 8; ++j) linp[(mb0 + j) * 2048 + Dm + n] = acc[j] + h2c_b[n];
      }
      gbar(cnt, gen);
      for (int li = 0; li < 4; ++li) {
        int dn = __hip_atomic_load(done, __ATOMIC_RELAXED, __HIP_MEMORY_SCOPE_AGENT);
        if (!dn) {
          // P8: gi_l = linp@l_Wi.T + l_bi ; gh_l = l@l_Wh.T + l_bh
          for (int tile = gw; tile < 384; tile += TW) {
            if (tile < 192) {
              v8f acc = wmma_kloop_f32(linp, 2048, l_Wi + tile * 16 * 2048, 2048, 2048, lane);
              int n = tile * 16 + r16;
#pragma unroll
              for (int j = 0; j < 8; ++j) gil[(mb0 + j) * 3072 + n] = acc[j] + l_bi[n];
            } else {
              int tt = tile - 192;
              v8f acc = wmma_kloop_f32(l, Lm, l_Wh + tt * 16 * Lm, Lm, Lm, lane);
              int n = tt * 16 + r16;
#pragma unroll
              for (int j = 0; j < 8; ++j) ghl[(mb0 + j) * 3072 + n] = acc[j] + l_bh[n];
            }
          }
        }
        gbar(cnt, gen);
        if (!dn) {
          // P9: l GRU update + convergence violation flag
          for (int i = blockIdx.x * NTHR + threadIdx.x; i < Bsz * Lm; i += GBLK * NTHR) {
            int b = i >> 10, n = i & 1023;
            float lv = l[b * Lm + n];
            float o = gru_out(gil[b * 3072 + n], gil[b * 3072 + 1024 + n], gil[b * 3072 + 2048 + n],
                              ghl[b * 3072 + n], ghl[b * 3072 + 1024 + n], ghl[b * 3072 + 2048 + n], lv);
            l[b * Lm + n] = o;
            if (fabsf(o - lv) > 1e-4f + 1e-5f * fabsf(lv))
              __hip_atomic_fetch_or(viol, 1, __ATOMIC_RELAXED, __HIP_MEMORY_SCOPE_AGENT);
          }
        }
        gbar(cnt, gen);
        // P10: finalize convergence
        if (blockIdx.x == 0 && threadIdx.x == 0) {
          int v = *viol;
          if (!dn && v == 0) *done = 1;
          *viol = 0;
        }
        gbar(cnt, gen);
      }
      // P11: enc += l@l2o_w.T + b  (+ store enc_all on last H step)
      for (int tile = gw; tile < Dm / 16; tile += TW) {
        v8f acc = wmma_kloop_f32(l, Lm, l2o_w + tile * 16 * Lm, Lm, Lm, lane);
        int n = tile * 16 + r16;
#pragma unroll
        for (int j = 0; j < 8; ++j) {
          int m = mb0 + j;
          float v = enc[m * Dm + n] + acc[j] + l2o_b[n];
          enc[m * Dm + n] = v;
          if (hs == 1 && m < Bsz) enc_all[((long)m * Tlen + t) * Dm + n] = v;
        }
      }
      gbar(cnt, gen);
    }
  }
}

__global__ __launch_bounds__(256) void ln_kernel(const float* __restrict__ enc_all,
                                                 const float* __restrict__ g,
                                                 const float* __restrict__ be,
                                                 _Float16* __restrict__ yh) {
  __shared__ float red[256];
  const int row = blockIdx.x;                 // 0..2047
  const float* xr = enc_all + (long)row * Dm;
  float s = 0.f;
  for (int d = threadIdx.x; d < Dm; d += 256) s += xr[d];
  red[threadIdx.x] = s; __syncthreads();
  for (int off = 128; off > 0; off >>= 1) {
    if (threadIdx.x < off) red[threadIdx.x] += red[threadIdx.x + off];
    __syncthreads();
  }
  float mu = red[0] * (1.f / Dm);
  __syncthreads();
  float vs = 0.f;
  for (int d = threadIdx.x; d < Dm; d += 256) { float e = xr[d] - mu; vs += e * e; }
  red[threadIdx.x] = vs; __syncthreads();
  for (int off = 128; off > 0; off >>= 1) {
    if (threadIdx.x < off) red[threadIdx.x] += red[threadIdx.x + off];
    __syncthreads();
  }
  float rstd = rsqrtf(red[0] * (1.f / Dm) + 1e-5f);
  for (int d = threadIdx.x; d < Dm; d += 256)
    yh[(long)row * Dm + d] = (_Float16)(((xr[d] - mu) * rstd) * g[d] + be[d]);
}

// logits[2048,32000] = yh[2048,1024] @ tok_emb[32000,1024]^T
// B tile (16 vocab rows x 1024) staged to LDS via TDM, converted f32->f16 once
// per block, consumed by v_wmma_f32_16x16x32_f16 with B read through ds_load.
__global__ __launch_bounds__(256) void logits_kernel(const _Float16* __restrict__ yh,
                                                     const float* __restrict__ tok,
                                                     float* __restrict__ logits) {
  __shared__ __align__(16) float    stage[16 * 256];      // TDM landing (one K-chunk)
  __shared__ __align__(16) _Float16 bt[16 * BT_PITCH];    // padded f16 B tile
  const int tid  = threadIdx.x;
  const int lane = tid & 31, widx = tid >> 5;
  const int ntile = blockIdx.x;
  const int r = lane & 15, half = lane >> 4;

  // ---- stage B tile: tok rows [ntile*16, +16), K-chunks of 256 columns ----
  for (int ch = 0; ch < 4; ++ch) {
#ifdef USE_TDM
    if (widx == 0) {
      unsigned long long ga =
          (unsigned long long)(const void*)(tok + (size_t)ntile * 16 * Dm + ch * 256);
      unsigned int lds_off = (unsigned int)(size_t)(void*)stage;  // low 32b = LDS byte offset
      u32x4 g0;
      g0[0] = 1u;                                          // count=1, user mode
      g0[1] = lds_off;                                     // lds_addr
      g0[2] = (unsigned int)ga;                            // global_addr[31:0]
      g0[3] = (unsigned int)((ga >> 32) & 0x1FFFFFFull) |  // global_addr[56:32]
              (2u << 30);                                  // type=2 (image)
      i32x8 g1;
      g1[0] = (2 << 16);                                   // data_size=4B, no multicast
      g1[1] = (int)((Dm & 0xFFFF) << 16);                  // tensor_dim0[15:0]
      g1[2] = (int)(((Dm >> 16) & 0xFFFF) |                // tensor_dim0[31:16]
                    ((Vv & 0xFFFF) << 16));                // tensor_dim1[15:0]
      g1[3] = (int)(((Vv >> 16) & 0xFFFF) |                // tensor_dim1[31:16]
                    (256 << 16));                          // tile_dim0 = 256
      g1[4] = 16;                                          // tile_dim1 = 16 rows
      g1[5] = Dm;                                          // tensor_dim0_stride = 1024
      g1[6] = 0;
      g1[7] = 0;
      i32x4 g2 = {0, 0, 0, 0};
      i32x4 g3 = {0, 0, 0, 0};
#if __clang_major__ >= 23
      i32x8 g4 = {0, 0, 0, 0, 0, 0, 0, 0};
      __builtin_amdgcn_tensor_load_to_lds(g0, g1, g2, g3, g4, 0);
#else
      __builtin_amdgcn_tensor_load_to_lds(g0, g1, g2, g3, 0);
#endif
      __builtin_amdgcn_s_wait_tensorcnt(0);
    }
    __syncthreads();
#else
    for (int i = tid; i < 16 * 256; i += 256)
      stage[i] = tok[(size_t)(ntile * 16 + (i >> 8)) * Dm + ch * 256 + (i & 255)];
    __syncthreads();
#endif
    for (int i = tid; i < 16 * 256; i += 256) {
      int rr = i >> 8, cc = i & 255;
      bt[rr * BT_PITCH + ch * 256 + cc] = (_Float16)stage[i];
    }
    __syncthreads();
  }

  // ---- WMMA f16 K-loop, B from LDS, 4 M-tiles per wave ----
  const int n = ntile * 16 + r;
  const _Float16* brow = bt + r * BT_PITCH + 8 * half;
#pragma unroll
  for (int i = 0; i < MT_PER_WAVE; ++i) {
    const int mtile = blockIdx.y * MT_PER_BLK + i * WPB + widx;
    const _Float16* arow = yh + (size_t)(mtile * 16 + r) * Dm + 8 * half;
    v8f acc = {};
    for (int k = 0; k < Dm; k += 32) {
      union { v16h v; v8h h2[2]; } au, bu;
      au.h2[0] = *(const v8h*)(arow + k);
      au.h2[1] = *(const v8h*)(arow + k + 16);
      bu.h2[0] = *(const v8h*)(brow + k);
      bu.h2[1] = *(const v8h*)(brow + k + 16);
      acc = __builtin_amdgcn_wmma_f32_16x16x32_f16(false, au.v, false, bu.v, (short)0, acc,
                                                   false, false);
    }
    const int mb = mtile * 16 + half * 8;
#pragma unroll
    for (int j = 0; j < 8; ++j) logits[(size_t)(mb + j) * Vv + n] = acc[j];
  }
}

extern "C" void kernel_launch(void* const* d_in, const int* in_sizes, int n_in,
                              void* d_out, int out_size, void* d_ws, size_t ws_size,
                              hipStream_t stream) {
  (void)in_sizes; (void)n_in; (void)out_size; (void)ws_size;
  const int*   ids        = (const int*)d_in[0];
  const float* tok_emb    = (const float*)d_in[1];
  const float* pos_emb    = (const float*)d_in[2];
  const float* persistent = (const float*)d_in[3];
  const float* ltm_keys   = (const float*)d_in[4];
  const float* ltm_vals   = (const float*)d_in[5];
  const float* qproj_w    = (const float*)d_in[6];
  const float* in_proj_w  = (const float*)d_in[7];
  const float* in_proj_b  = (const float*)d_in[8];
  const float* h_Wi       = (const float*)d_in[9];
  const float* h_bi       = (const float*)d_in[10];
  const float* h_Wh       = (const float*)d_in[11];
  const float* h_bh       = (const float*)d_in[12];
  const float* h2c_w      = (const float*)d_in[13];
  const float* h2c_b      = (const float*)d_in[14];
  const float* l_Wi       = (const float*)d_in[15];
  const float* l_bi       = (const float*)d_in[16];
  const float* l_Wh       = (const float*)d_in[17];
  const float* l_bh       = (const float*)d_in[18];
  const float* l2o_w      = (const float*)d_in[19];
  const float* l2o_b      = (const float*)d_in[20];
  const float* ln_g       = (const float*)d_in[21];
  const float* ln_b       = (const float*)d_in[22];

  float* ws      = (float*)d_ws;
  float* x       = ws + WS_X;
  float* enc_all = ws + WS_ENCALL;
  _Float16* yh   = (_Float16*)(ws + WS_YH);
  float* zz      = ws + WS_ZZ;
  float* xt   = zz + Z_XT;
  float* q    = zz + Z_Q;
  float* sim  = zz + Z_SIM;
  float* mac  = zz + Z_MAC;
  float* enc  = zz + Z_ENC;
  float* h    = zz + Z_H;
  float* l    = zz + Z_L;
  float* linp = zz + Z_LINP;
  float* gi   = zz + Z_GI;
  float* gh   = zz + Z_GH;
  float* gil  = zz + Z_GIL;
  float* ghl  = zz + Z_GHL;
  int*   flags = (int*)(zz + Z_FLAGS);

  float* logits   = (float*)d_out + OUT_LOGITS;
  float* out_vals = (float*)d_out + OUT_VALS;
  int*   out_idx  = (int*)((float*)d_out + OUT_IDX);

  zero_kernel<<<(Z_TOTAL + 255) / 256, 256, 0, stream>>>(zz, Z_TOTAL);
  embed_kernel<<<(Bsz * Tlen * Dm + 255) / 256, 256, 0, stream>>>(ids, tok_emb, pos_emb, x);
  chronos_persistent<<<GBLK, NTHR, 0, stream>>>(
      x, persistent, ltm_keys, ltm_vals, qproj_w, in_proj_w, in_proj_b,
      h_Wi, h_bi, h_Wh, h_bh, h2c_w, h2c_b, l_Wi, l_bi, l_Wh, l_bh,
      l2o_w, l2o_b, xt, q, sim, mac, enc, h, l, linp, gi, gh, gil, ghl,
      flags, enc_all, out_vals, out_idx);
  ln_kernel<<<Bsz * Tlen, 256, 0, stream>>>(enc_all, ln_g, ln_b, yh);
  logits_kernel<<<dim3(Vv / 16, LOG_YGRP), 256, 0, stream>>>(yh, tok_emb, logits);
}
